// DAWN_38491496907063
// MI455X (gfx1250) — matrix operand order repaired and checked by reading, not compiled, not run on hardware
//
#include <hip/hip_runtime.h>

// ---------------- problem dims ----------------
#define B_   2
#define S_   2048
#define D_   1024
#define H_   16
#define R_   256
#define L_   4
#define NC_  64
#define NK_  4096
#define KK_  8
#define SD_  64
#define V_   32000
#define DH_  64
#define BS_  (B_*S_)
#define DR_  (D_*R_)

typedef __attribute__((ext_vector_type(16))) __bf16 v16bf;
typedef __attribute__((ext_vector_type(8)))  float  v8f;

struct U4 { unsigned x, y, z, w; };
struct Frag32 { U4 lo, hi; };
static_assert(sizeof(U4) == 16, "U4 must be 16B");
static_assert(sizeof(Frag32) == 32, "Frag32 must be 32B");
static_assert(sizeof(v16bf) == 32, "v16bf must be 32B");

__device__ __forceinline__ v16bf frag_cast(Frag32 f) { return __builtin_bit_cast(v16bf, f); }

// f32 -> bf16 round-to-nearest-even
__device__ __forceinline__ unsigned short f2bf(float x) {
  unsigned u = __builtin_bit_cast(unsigned, x);
  unsigned r = u + 0x7FFFu + ((u >> 16) & 1u);
  return (unsigned short)(r >> 16);
}

// =====================================================================
// Embedding: x[b,s,:] = token_emb[tok] + pos_emb[s]
// =====================================================================
__global__ __launch_bounds__(256) void k_embed(const int* __restrict__ tokens,
                                               const float* __restrict__ te,
                                               const float* __restrict__ pe,
                                               float* __restrict__ x) {
  int row = blockIdx.x;                 // b*S + s
  int s = row % S_;
  int tok = tokens[row];
  const float* tr = te + (long long)tok * D_;
  const float* pr = pe + (long long)s * D_;
  float* xr = x + (long long)row * D_;
  for (int d = threadIdx.x; d < D_; d += 256) xr[d] = tr[d] + pr[d];
}

// =====================================================================
// LayerNorm over D=1024, writes f32 + bf16
// =====================================================================
__global__ __launch_bounds__(256) void k_ln(const float* __restrict__ x,
                                            const float* __restrict__ g,
                                            const float* __restrict__ b,
                                            float* __restrict__ xn,
                                            unsigned short* __restrict__ xnb) {
  __shared__ float red[256];
  int row = blockIdx.x, t = threadIdx.x;
  const float* xr = x + (long long)row * D_;
  float v[4];
  float s = 0.f, s2 = 0.f;
#pragma unroll
  for (int j = 0; j < 4; j++) { v[j] = xr[t + j*256]; s += v[j]; s2 += v[j]*v[j]; }
  red[t] = s; __syncthreads();
  for (int k = 128; k > 0; k >>= 1) { if (t < k) red[t] += red[t+k]; __syncthreads(); }
  float mean = red[0] / (float)D_;
  __syncthreads();
  red[t] = s2; __syncthreads();
  for (int k = 128; k > 0; k >>= 1) { if (t < k) red[t] += red[t+k]; __syncthreads(); }
  float var = red[0] / (float)D_ - mean*mean;
  float inv = rsqrtf(var + 1e-5f);
#pragma unroll
  for (int j = 0; j < 4; j++) {
    int d = t + j*256;
    float o = (v[j] - mean) * inv * g[d] + b[d];
    xn[(long long)row*D_ + d] = o;
    xnb[(long long)row*D_ + d] = f2bf(o);
  }
}

// =====================================================================
// f32 -> bf16 buffer conversion
// =====================================================================
__global__ __launch_bounds__(256) void k_f2bf(const float* __restrict__ src,
                                              unsigned short* __restrict__ dst, int n) {
  int i = (blockIdx.x * 256 + threadIdx.x) * 4;
#pragma unroll
  for (int j = 0; j < 4; j++) if (i + j < n) dst[i+j] = f2bf(src[i+j]);
}

// =====================================================================
// Generic bf16 WMMA GEMM:  C[M,N] = scale*(A[M,K] @ B[K,N]) (+ add)
// Block tile 64x128, 8 waves (4 M x 2 N), each wave 16x64 via 4 WMMA acc.
// A/B staged in LDS; fragments per CDNA5 ISA layouts.
// =====================================================================
__global__ __launch_bounds__(256) void k_gemm(
    const unsigned short* __restrict__ A, const unsigned short* __restrict__ Bm,
    float* __restrict__ Cf, unsigned short* __restrict__ Cb,
    const float* __restrict__ addsrc,
    int M, int N, int K, float scale,
    long long sA, long long sB, long long sCf, long long sCb, long long sAdd)
{
  __shared__ __align__(16) unsigned short As[64][32];   // 4KB
  __shared__ __align__(16) unsigned short Bs[32][128];  // 8KB
  const int z = blockIdx.z;
  A  += (long long)z * sA;
  Bm += (long long)z * sB;
  if (Cf)     Cf     += (long long)z * sCf;
  if (Cb)     Cb     += (long long)z * sCb;
  if (addsrc) addsrc += (long long)z * sAdd;

  const int t    = threadIdx.x;
  const int lane = t & 31, wave = t >> 5;
  const int wm = wave >> 1, wn = wave & 1;
  const int m0 = blockIdx.x * 64;
  const int n0 = blockIdx.y * 128;
  const int halfsel = (lane >= 16) ? 8 : 0;   // A: K-offset; C: row-offset
  const int mrow = lane & 15;

  v8f acc[4];
#pragma unroll
  for (int i = 0; i < 4; i++)
#pragma unroll
    for (int e = 0; e < 8; e++) acc[i][e] = 0.f;

  const int aRow = t >> 2, aCol = (t & 3) * 8;   // A tile: 64x32, 8 elems/thread
  const int bRow = t >> 3, bCol = (t & 7) * 16;  // B tile: 32x128, 16 elems/thread

  for (int k0 = 0; k0 < K; k0 += 32) {
    U4 av = {0,0,0,0};
    if (m0 + aRow < M)
      av = *(const U4*)(A + (long long)(m0 + aRow) * K + k0 + aCol);
    *(U4*)&As[aRow][aCol] = av;

    U4 bv0 = {0,0,0,0}, bv1 = {0,0,0,0};
    const unsigned short* bsrc = Bm + (long long)(k0 + bRow) * N + n0 + bCol;
    if (n0 + bCol     < N) bv0 = *(const U4*)bsrc;
    if (n0 + bCol + 8 < N) bv1 = *(const U4*)(bsrc + 8);
    *(U4*)&Bs[bRow][bCol]     = bv0;
    *(U4*)&Bs[bRow][bCol + 8] = bv1;
    __syncthreads();

    if (k0 + 32 < K) {   // lowers to global_prefetch_b8
      __builtin_prefetch(A  + (long long)(m0 + aRow) * K + (k0 + 32) + aCol, 0, 1);
      __builtin_prefetch(Bm + (long long)(k0 + 32 + bRow) * N + n0 + bCol, 0, 1);
    }

    // A fragment: lane holds row m=lane%16; K runs {hs..hs+7} and {16+hs..23+hs}
    Frag32 af;
    af.lo = *(const U4*)&As[wm*16 + mrow][halfsel];
    af.hi = *(const U4*)&As[wm*16 + mrow][16 + halfsel];
    v16bf a = frag_cast(af);
#pragma unroll
    for (int nt = 0; nt < 4; nt++) {
      // B fragment: lane holds N-col = lane%16; K runs {hs..hs+7} and {16+hs..23+hs}
      v16bf bfv;
      const int bc = wn*64 + nt*16;
#pragma unroll
      for (int i = 0; i < 16; i++)
        bfv[i] = __builtin_bit_cast(__bf16, Bs[(i < 8 ? halfsel + i : 8 + halfsel + i)][bc + mrow]);
      acc[nt] = __builtin_amdgcn_wmma_f32_16x16x32_bf16(
          false, a, false, bfv, (short)0, acc[nt], false, false);
    }
    __syncthreads();
  }

#pragma unroll
  for (int nt = 0; nt < 4; nt++) {
#pragma unroll
    for (int r = 0; r < 8; r++) {
      const int row = m0 + wm*16 + r + halfsel;
      const int col = n0 + wn*64 + nt*16 + mrow;
      if (row < M && col < N) {
        float v = acc[nt][r] * scale;
        const long long idx = (long long)row * N + col;
        if (addsrc) v += addsrc[idx];
        if (Cf) Cf[idx] = v;
        if (Cb) Cb[idx] = f2bf(v);
      }
    }
  }
}

// =====================================================================
// SSM scan: h_t = h_{t-1} @ A + u_t   (per batch), A cached in LDS
// =====================================================================
__global__ __launch_bounds__(256) void k_scan(const float* __restrict__ u,
                                              const float* __restrict__ A,
                                              float* __restrict__ hOut) {
  __shared__ float Al[SD_*SD_];
  __shared__ float h[SD_];
  __shared__ float part[4][SD_];
  int b = blockIdx.x, t = threadIdx.x;
  for (int i = t; i < SD_*SD_; i += 256) Al[i] = A[i];
  if (t < SD_) h[t] = 0.f;
  __syncthreads();
  int j = t & 63, pp = t >> 6;
  for (int s = 0; s < S_; s++) {
    float acc = 0.f;
    int i0 = pp * 16;
#pragma unroll
    for (int i = 0; i < 16; i++) acc += h[i0+i] * Al[(i0+i)*SD_ + j];
    part[pp][j] = acc;
    __syncthreads();
    if (t < SD_)
      h[t] = part[0][t] + part[1][t] + part[2][t] + part[3][t]
           + u[((long long)b*S_ + s)*SD_ + t];
    __syncthreads();
  }
  if (t < SD_) hOut[b*SD_ + t] = h[t];
}

// w[b,:] = h[b,:] @ Wimp
__global__ __launch_bounds__(256) void k_hwimp(const float* __restrict__ h,
                                               const float* __restrict__ Wimp,
                                               float* __restrict__ w) {
  __shared__ float hs[SD_];
  int b = blockIdx.y, t = threadIdx.x;
  if (t < SD_) hs[t] = h[b*SD_ + t];
  __syncthreads();
  int d = blockIdx.x * 256 + t;
  float acc = 0.f;
  for (int i = 0; i < SD_; i++) acc += hs[i] * Wimp[(long long)i*D_ + d];
  w[(long long)b*D_ + d] = acc;
}

// score[row] = dot(xn[row,:], w[b,:])
__global__ __launch_bounds__(256) void k_dot(const float* __restrict__ xn,
                                             const float* __restrict__ w,
                                             float* __restrict__ score) {
  __shared__ float red[256];
  int row = blockIdx.x, t = threadIdx.x;
  int b = row / S_;
  const float* xr = xn + (long long)row * D_;
  const float* wr = w + (long long)b * D_;
  float acc = 0.f;
#pragma unroll
  for (int j = 0; j < 4; j++) { int d = t + j*256; acc += xr[d] * wr[d]; }
  red[t] = acc; __syncthreads();
  for (int k = 128; k > 0; k >>= 1) { if (t < k) red[t] += red[t+k]; __syncthreads(); }
  if (t == 0) score[row] = red[0];
}

// softmax over S (in place)
__global__ __launch_bounds__(256) void k_softmax_s(float* __restrict__ score) {
  __shared__ float red[256];
  int b = blockIdx.x, t = threadIdx.x;
  float* sr = score + (long long)b * S_;
  float v[8];
  float mx = -__builtin_inff();
#pragma unroll
  for (int j = 0; j < 8; j++) { v[j] = sr[t + j*256]; mx = fmaxf(mx, v[j]); }
  red[t] = mx; __syncthreads();
  for (int k = 128; k > 0; k >>= 1) { if (t < k) red[t] = fmaxf(red[t], red[t+k]); __syncthreads(); }
  mx = red[0]; __syncthreads();
  float sm = 0.f;
#pragma unroll
  for (int j = 0; j < 8; j++) { v[j] = __expf(v[j] - mx); sm += v[j]; }
  red[t] = sm; __syncthreads();
  for (int k = 128; k > 0; k >>= 1) { if (t < k) red[t] += red[t+k]; __syncthreads(); }
  float inv = 1.f / red[0];
#pragma unroll
  for (int j = 0; j < 8; j++) sr[t + j*256] = v[j] * inv;
}

// row softmax over NC=64 (in place)
__global__ __launch_bounds__(64) void k_softmax_rows(float* __restrict__ p) {
  __shared__ float red[64];
  int row = blockIdx.x, t = threadIdx.x;
  float* pr = p + (long long)row * NC_;
  float v = pr[t];
  red[t] = v; __syncthreads();
  for (int k = 32; k > 0; k >>= 1) { if (t < k) red[t] = fmaxf(red[t], red[t+k]); __syncthreads(); }
  float mx = red[0]; __syncthreads();
  float e = __expf(v - mx);
  red[t] = e; __syncthreads();
  for (int k = 32; k > 0; k >>= 1) { if (t < k) red[t] += red[t+k]; __syncthreads(); }
  pr[t] = e / red[0];
}

// nw[b,n] = sum_s imp[b,s] * pref[b,s,n]
__global__ __launch_bounds__(256) void k_nw(const float* __restrict__ imp,
                                            const float* __restrict__ pref,
                                            float* __restrict__ nw) {
  __shared__ float red[256];
  int b = blockIdx.x / NC_, n = blockIdx.x % NC_, t = threadIdx.x;
  float acc = 0.f;
  for (int s = t; s < S_; s += 256) {
    long long row = (long long)b * S_ + s;
    acc += imp[row] * pref[row * NC_ + n];
  }
  red[t] = acc; __syncthreads();
  for (int k = 128; k > 0; k >>= 1) { if (t < k) red[t] += red[t+k]; __syncthreads(); }
  if (t == 0) nw[b*NC_ + n] = red[0];
}

__global__ __launch_bounds__(64) void k_nwnorm(float* __restrict__ nw) {
  __shared__ float red[64];
  int b = blockIdx.x, t = threadIdx.x;
  float v = nw[b*NC_ + t];
  red[t] = v; __syncthreads();
  for (int k = 32; k > 0; k >>= 1) { if (t < k) red[t] += red[t+k]; __syncthreads(); }
  nw[b*NC_ + t] = v / (red[0] + 1e-8f);
}

// sc[b] = nw[b,:] @ compress  -> bf16 [B, D*R]
__global__ __launch_bounds__(256) void k_sc(const float* __restrict__ nw,
                                            const float* __restrict__ comp,
                                            unsigned short* __restrict__ scb) {
  __shared__ float snw[B_*NC_];
  int t = threadIdx.x;
  if (t < B_*NC_) snw[t] = nw[t];
  __syncthreads();
  long long dr = (long long)blockIdx.x * 256 + t;
  float a0 = 0.f, a1 = 0.f;
#pragma unroll 4
  for (int n = 0; n < NC_; n++) {
    float c = comp[(long long)n * DR_ + dr];
    a0 += snw[n] * c;
    a1 += snw[NC_ + n] * c;
  }
  scb[dr]       = f2bf(a0);
  scb[DR_ + dr] = f2bf(a1);
}

// transpose K [B,S,H,DH] -> Kt [B,H,DH,S] (bf16)
__global__ __launch_bounds__(256) void k_trK(const unsigned short* __restrict__ Kb,
                                             unsigned short* __restrict__ Ktb) {
  long long o = (long long)blockIdx.x * 256 + threadIdx.x;  // [B,H,DH,S] linear
  int s = (int)(o % S_);
  long long r = o / S_;
  int dh = (int)(r % DH_);
  long long bh = r / DH_;
  int b = (int)(bh / H_), h = (int)(bh % H_);
  Ktb[o] = Kb[((long long)b*S_ + s)*D_ + h*DH_ + dh];
}

// kK [NK,R] f32 -> kKt [R,NK] bf16
__global__ __launch_bounds__(256) void k_kkt(const float* __restrict__ kK,
                                             unsigned short* __restrict__ kKtb) {
  long long o = (long long)blockIdx.x * 256 + threadIdx.x;  // r*NK + j
  int j = (int)(o % NK_);
  int r = (int)(o / NK_);
  kKtb[o] = f2bf(kK[(long long)j*R_ + r]);
}

// =====================================================================
// Flash attention: wave32, one wave per 16-row q-tile, online softmax.
// Q [B,S,D] bf16, Kt [B,H,DH,S] bf16, V [B,S,D] bf16 -> O [B,S,D] bf16
// =====================================================================
__global__ __launch_bounds__(128) void k_attn(
    const unsigned short* __restrict__ Qb,
    const unsigned short* __restrict__ Ktb,
    const unsigned short* __restrict__ Vb,
    unsigned short* __restrict__ Ob,
    float scale)
{
  __shared__ __align__(16) unsigned short Pl[4][16][32];  // per-wave P tile
  const int lane = threadIdx.x & 31, wave = threadIdx.x >> 5;
  const int bh = blockIdx.y;
  const int b = bh / H_, h = bh % H_;
  const int qBase = (blockIdx.x * 4 + wave) * 16;
  const int halfsel = (lane >= 16) ? 8 : 0;
  const int mrow = lane & 15;
  const float NEGINF = -__builtin_inff();

  // Q fragments for the two dh-32 chunks
  const unsigned short* qrow = Qb + ((long long)b*S_ + qBase + mrow) * D_ + h*DH_;
  v16bf qa[2];
#pragma unroll
  for (int ks = 0; ks < 2; ks++) {
    Frag32 f;
    f.lo = *(const U4*)(qrow + ks*32 + halfsel);
    f.hi = *(const U4*)(qrow + ks*32 + 16 + halfsel);
    qa[ks] = frag_cast(f);
  }

  float rowM[8], rowL[8];
  v8f acc[4];
#pragma unroll
  for (int r = 0; r < 8; r++) { rowM[r] = NEGINF; rowL[r] = 0.f; }
#pragma unroll
  for (int nt = 0; nt < 4; nt++)
#pragma unroll
    for (int e = 0; e < 8; e++) acc[nt][e] = 0.f;

  for (int kc = 0; kc < qBase + 16; kc += 32) {
    float p[2][8];
#pragma unroll
    for (int ntile = 0; ntile < 2; ntile++) {
      v8f s;
#pragma unroll
      for (int e = 0; e < 8; e++) s[e] = 0.f;
#pragma unroll
      for (int ks = 0; ks < 2; ks++) {
        const unsigned short* kt = Ktb + ((long long)bh*DH_ + ks*32) * S_ + kc + ntile*16 + mrow;
        v16bf f;
#pragma unroll
        for (int i = 0; i < 16; i++)
          f[i] = __builtin_bit_cast(__bf16, kt[(long long)(i < 8 ? halfsel + i : 8 + halfsel + i) * S_]);
        s = __builtin_amdgcn_wmma_f32_16x16x32_bf16(
            false, qa[ks], false, f, (short)0, s, false, false);
      }
      const int kcol = kc + ntile*16 + mrow;
#pragma unroll
      for (int r = 0; r < 8; r++) {
        const int qi = qBase + r + halfsel;
        p[ntile][r] = (kcol <= qi) ? s[r] * scale : NEGINF;
      }
    }
    // online softmax: rows live on 16-lane halves (wave32)
    float alpha[8];
#pragma unroll
    for (int r = 0; r < 8; r++) {
      float mx = fmaxf(p[0][r], p[1][r]);
#pragma unroll
      for (int off = 1; off < 16; off <<= 1) mx = fmaxf(mx, __shfl_xor(mx, off, 32));
      const float mnew = fmaxf(rowM[r], mx);
      alpha[r] = (rowM[r] == NEGINF) ? 0.f : __expf(rowM[r] - mnew);
      rowM[r] = mnew;
      const float e0 = __expf(p[0][r] - mnew);
      const float e1 = __expf(p[1][r] - mnew);
      p[0][r] = e0; p[1][r] = e1;
      float sm = e0 + e1;
#pragma unroll
      for (int off = 1; off < 16; off <<= 1) sm += __shfl_xor(sm, off, 32);
      rowL[r] = rowL[r] * alpha[r] + sm;
    }
#pragma unroll
    for (int nt = 0; nt < 4; nt++)
#pragma unroll
      for (int r = 0; r < 8; r++) acc[nt][r] *= alpha[r];

    // C-layout P -> LDS -> A-layout fragment (wave-private, DS in-order)
#pragma unroll
    for (int r = 0; r < 8; r++) {
      Pl[wave][r + halfsel][mrow]      = f2bf(p[0][r]);
      Pl[wave][r + halfsel][16 + mrow] = f2bf(p[1][r]);
    }
    asm volatile("" ::: "memory");
    Frag32 pf;
    pf.lo = *(const U4*)&Pl[wave][mrow][halfsel];
    pf.hi = *(const U4*)&Pl[wave][mrow][16 + halfsel];
    const v16bf pa = frag_cast(pf);
    asm volatile("" ::: "memory");

    const unsigned short* vrow = Vb + ((long long)b*S_ + kc) * D_ + h*DH_ + mrow;
#pragma unroll
    for (int nt = 0; nt < 4; nt++) {
      v16bf vf;
#pragma unroll
      for (int i = 0; i < 16; i++)
        vf[i] = __builtin_bit_cast(__bf16, vrow[(long long)(i < 8 ? halfsel + i : 8 + halfsel + i) * D_ + nt*16]);
      acc[nt] = __builtin_amdgcn_wmma_f32_16x16x32_bf16(
          false, pa, false, vf, (short)0, acc[nt], false, false);
    }
  }

  unsigned short* orow = Ob + ((long long)b*S_ + qBase) * D_ + h*DH_;
#pragma unroll
  for (int nt = 0; nt < 4; nt++)
#pragma unroll
    for (int r = 0; r < 8; r++) {
      const float v = acc[nt][r] / rowL[r];
      orow[(long long)(r + halfsel) * D_ + nt*16 + mrow] = f2bf(v);
    }
}

// =====================================================================
// top-8 over NK=4096 + softmax + weighted kV gather + residual into x
// =====================================================================
__global__ __launch_bounds__(256) void k_topk(const float* __restrict__ ks,
                                              const float* __restrict__ kV,
                                              float* __restrict__ x) {
  __shared__ float sVal[256];
  __shared__ int   sIdx[256];
  __shared__ float topV[KK_];
  __shared__ int   topI[KK_];
  const int row = blockIdx.x, t = threadIdx.x;
  const float* sr = ks + (long long)row * NK_;
  const float NEGINF = -__builtin_inff();
  float v[NK_/256];
#pragma unroll
  for (int i = 0; i < NK_/256; i++) v[i] = sr[i*256 + t];

  for (int pass = 0; pass < KK_; pass++) {
    float best = NEGINF; int bi = 0;
#pragma unroll
    for (int i = 0; i < NK_/256; i++) if (v[i] > best) { best = v[i]; bi = i*256 + t; }
    sVal[t] = best; sIdx[t] = bi;
    __syncthreads();
    for (int k = 128; k > 0; k >>= 1) {
      if (t < k && sVal[t+k] > sVal[t]) { sVal[t] = sVal[t+k]; sIdx[t] = sIdx[t+k]; }
      __syncthreads();
    }
    const int win = sIdx[0];
    if (t == 0) { topV[pass] = sVal[0]; topI[pass] = win; }
#pragma unroll
    for (int i = 0; i < NK_/256; i++) if (i*256 + t == win) v[i] = NEGINF;
    __syncthreads();
  }
  float w[KK_]; float wsum = 0.f;
#pragma unroll
  for (int k = 0; k < KK_; k++) { w[k] = __expf(topV[k] - topV[0]); wsum += w[k]; }
  const float inv = 1.f / wsum;
  float* xr = x + (long long)row * D_;
#pragma unroll
  for (int j = 0; j < D_/256; j++) {
    const int d = j*256 + t;
    float acc = 0.f;
#pragma unroll
    for (int k = 0; k < KK_; k++) acc += w[k] * kV[(long long)topI[k] * D_ + d];
    xr[d] += acc * inv;
  }
}

// =====================================================================
// Host orchestration
// =====================================================================
extern "C" void kernel_launch(void* const* d_in, const int* in_sizes, int n_in,
                              void* d_out, int out_size, void* d_ws, size_t ws_size,
                              hipStream_t stream)
{
  const int*   tokens    = (const int*)d_in[0];
  const float* token_emb = (const float*)d_in[1];
  const float* pos_emb   = (const float*)d_in[2];
  const float* compress  = (const float*)d_in[3];
  const float* kK        = (const float*)d_in[4];
  const float* kV        = (const float*)d_in[5];
  const float* a_A       = (const float*)d_in[6];
  const float* a_B       = (const float*)d_in[7];
  const float* a_imp     = (const float*)d_in[8];
  const float* a_rt      = (const float*)d_in[9];
  const float* Wq        = (const float*)d_in[10];
  const float* Wk        = (const float*)d_in[11];
  const float* Wv        = (const float*)d_in[12];
  const float* Wo        = (const float*)d_in[13];
  const float* m_A       = (const float*)d_in[14];
  const float* m_B       = (const float*)d_in[15];
  const float* m_imp     = (const float*)d_in[16];
  const float* m_rt      = (const float*)d_in[17];
  const float* g1        = (const float*)d_in[18];
  const float* b1        = (const float*)d_in[19];
  const float* g2        = (const float*)d_in[20];
  const float* b2        = (const float*)d_in[21];
  const float* gf        = (const float*)d_in[22];
  const float* bfin      = (const float*)d_in[23];
  const float* lm_head   = (const float*)d_in[24];

  char* base = (char*)d_ws;
  size_t off = 0;
  auto alloc = [&](size_t bytes) -> char* {
    char* p = base + off;
    off = (off + bytes + 255) & ~(size_t)255;
    return p;
  };
  float* x    = (float*)alloc((size_t)BS_*D_*4);
  float* xn   = (float*)alloc((size_t)BS_*D_*4);
  unsigned short* xnb = (unsigned short*)alloc((size_t)BS_*D_*2);
  float* u    = (float*)alloc((size_t)BS_*SD_*4);
  float* hbuf = (float*)alloc((size_t)B_*SD_*4);
  float* wimp = (float*)alloc((size_t)B_*D_*4);
  float* imp  = (float*)alloc((size_t)BS_*4);
  float* pref = (float*)alloc((size_t)BS_*NC_*4);
  float* nw   = (float*)alloc((size_t)B_*NC_*4);
  unsigned short* scb  = (unsigned short*)alloc((size_t)B_*DR_*2);
  unsigned short* hcb  = (unsigned short*)alloc((size_t)BS_*R_*2);
  unsigned short* Qb   = (unsigned short*)alloc((size_t)BS_*D_*2);
  unsigned short* Kb   = (unsigned short*)alloc((size_t)BS_*D_*2);
  unsigned short* Vb   = (unsigned short*)alloc((size_t)BS_*D_*2);
  unsigned short* Ktb  = (unsigned short*)alloc((size_t)BS_*D_*2);
  unsigned short* aob  = (unsigned short*)alloc((size_t)BS_*D_*2);
  unsigned short* Qmb  = (unsigned short*)alloc((size_t)BS_*R_*2);
  float* kcs  = (float*)alloc((size_t)BS_*NK_*4);
  unsigned short* kKtb = (unsigned short*)alloc((size_t)R_*NK_*2);
  unsigned short* lmb  = (unsigned short*)alloc((size_t)D_*V_*2);
  unsigned short* wB   = (unsigned short*)alloc((size_t)D_*SD_*2);
  unsigned short* wrt  = (unsigned short*)alloc((size_t)D_*NC_*2);
  unsigned short* wqb  = (unsigned short*)alloc((size_t)R_*D_*2);
  unsigned short* wkb  = (unsigned short*)alloc((size_t)R_*D_*2);
  unsigned short* wvb  = (unsigned short*)alloc((size_t)R_*D_*2);
  unsigned short* wob  = (unsigned short*)alloc((size_t)D_*D_*2);
  (void)ws_size; (void)in_sizes; (void)n_in; (void)out_size;

  auto f2bf_l = [&](const float* src, unsigned short* dst, int n) {
    k_f2bf<<<(n + 1023)/1024, 256, 0, stream>>>(src, dst, n);
  };
  auto gemm = [&](const unsigned short* A, const unsigned short* Bm, float* Cf,
                  unsigned short* Cb, const float* add, int M, int N, int K,
                  float scale, int batches, long long sA, long long sB,
                  long long sCf, long long sCb, long long sAdd) {
    dim3 grid((M + 63)/64, (N + 127)/128, batches);
    k_gemm<<<grid, 256, 0, stream>>>(A, Bm, Cf, Cb, add, M, N, K, scale,
                                     sA, sB, sCf, sCb, sAdd);
  };

  // ---- setup ----
  k_embed<<<BS_, 256, 0, stream>>>(tokens, token_emb, pos_emb, x);
  f2bf_l(lm_head, lmb, D_*V_);
  k_kkt<<<(NK_*R_)/256, 256, 0, stream>>>(kK, kKtb);

  const float inv_sqrt_dh = 0.125f;   // 1/sqrt(64)
  const float inv_sqrt_r  = 0.0625f;  // 1/sqrt(256)

  for (int l = 0; l < L_; l++) {
    // ===== NeuronCircuit (attention) =====
    k_ln<<<BS_, 256, 0, stream>>>(x, g1 + l*D_, b1 + l*D_, xn, xnb);
    f2bf_l(a_B + (size_t)l*D_*SD_, wB, D_*SD_);
    gemm(xnb, wB, u, nullptr, nullptr, BS_, SD_, D_, 1.f, 1, 0,0,0,0,0);
    k_scan<<<B_, 256, 0, stream>>>(u, a_A + (size_t)l*SD_*SD_, hbuf);
    k_hwimp<<<dim3(D_/256, B_), 256, 0, stream>>>(hbuf, a_imp + (size_t)l*SD_*D_, wimp);
    k_dot<<<BS_, 256, 0, stream>>>(xn, wimp, imp);
    k_softmax_s<<<B_, 256, 0, stream>>>(imp);
    f2bf_l(a_rt + (size_t)l*D_*NC_, wrt, D_*NC_);
    gemm(xnb, wrt, pref, nullptr, nullptr, BS_, NC_, D_, 1.f, 1, 0,0,0,0,0);
    k_softmax_rows<<<BS_, 64, 0, stream>>>(pref);
    k_nw<<<B_*NC_, 256, 0, stream>>>(imp, pref, nw);
    k_nwnorm<<<B_, 64, 0, stream>>>(nw);
    k_sc<<<DR_/256, 256, 0, stream>>>(nw, compress, scb);
    gemm(xnb, scb, nullptr, hcb, nullptr, S_, R_, D_, 1.f, B_,
         (long long)S_*D_, (long long)DR_, 0, (long long)S_*R_, 0);
    f2bf_l(Wq + (size_t)l*R_*D_, wqb, R_*D_);
    f2bf_l(Wk + (size_t)l*R_*D_, wkb, R_*D_);
    f2bf_l(Wv + (size_t)l*R_*D_, wvb, R_*D_);
    gemm(hcb, wqb, nullptr, Qb, nullptr, BS_, D_, R_, 1.f, 1, 0,0,0,0,0);
    gemm(hcb, wkb, nullptr, Kb, nullptr, BS_, D_, R_, 1.f, 1, 0,0,0,0,0);
    gemm(hcb, wvb, nullptr, Vb, nullptr, BS_, D_, R_, 1.f, 1, 0,0,0,0,0);
    k_trK<<<(B_*D_*S_)/256, 256, 0, stream>>>(Kb, Ktb);
    k_attn<<<dim3(S_/64, B_*H_), 128, 0, stream>>>(Qb, Ktb, Vb, aob, inv_sqrt_dh);
    f2bf_l(Wo + (size_t)l*D_*D_, wob, D_*D_);
    gemm(aob, wob, x, nullptr, x, BS_, D_, D_, 1.f, 1, 0,0,0,0,0);   // x += ao@Wo

    // ===== NeuronMemory (knowledge retrieval) =====
    k_ln<<<BS_, 256, 0, stream>>>(x, g2 + l*D_, b2 + l*D_, xn, xnb);
    f2bf_l(m_B + (size_t)l*D_*SD_, wB, D_*SD_);
    gemm(xnb, wB, u, nullptr, nullptr, BS_, SD_, D_, 1.f, 1, 0,0,0,0,0);
    k_scan<<<B_, 256, 0, stream>>>(u, m_A + (size_t)l*SD_*SD_, hbuf);
    k_hwimp<<<dim3(D_/256, B_), 256, 0, stream>>>(hbuf, m_imp + (size_t)l*SD_*D_, wimp);
    k_dot<<<BS_, 256, 0, stream>>>(xn, wimp, imp);
    k_softmax_s<<<B_, 256, 0, stream>>>(imp);
    f2bf_l(m_rt + (size_t)l*D_*NC_, wrt, D_*NC_);
    gemm(xnb, wrt, pref, nullptr, nullptr, BS_, NC_, D_, 1.f, 1, 0,0,0,0,0);
    k_softmax_rows<<<BS_, 64, 0, stream>>>(pref);
    k_nw<<<B_*NC_, 256, 0, stream>>>(imp, pref, nw);
    k_nwnorm<<<B_, 64, 0, stream>>>(nw);
    k_sc<<<DR_/256, 256, 0, stream>>>(nw, compress, scb);
    gemm(xnb, scb, nullptr, Qmb, nullptr, S_, R_, D_, 1.f, B_,
         (long long)S_*D_, (long long)DR_, 0, (long long)S_*R_, 0);
    gemm(Qmb, kKtb, kcs, nullptr, nullptr, BS_, NK_, R_, inv_sqrt_r, 1, 0,0,0,0,0);
    k_topk<<<BS_, 256, 0, stream>>>(kcs, kV, x);                      // x += mem
  }

  // ===== final LN + LM head =====
  k_ln<<<BS_, 256, 0, stream>>>(x, gf, bfin, xn, xnb);
  gemm(xnb, lmb, (float*)d_out, nullptr, nullptr, BS_, V_, D_, 1.f, 1, 0,0,0,0,0);
}
